// MultiHeadAttention_12326556139979
// MI455X (gfx1250) — matrix operand order repaired and checked by reading, not compiled, not run on hardware
//
#include <hip/hip_runtime.h>

// ---------------- problem constants ----------------
#define BATCH  32
#define SEQ    512
#define DMODEL 512
#define NHEAD  8
#define DEPTH  64
#define BS     (BATCH * SEQ)          // 16384 rows

// ---------------- vector types ----------------
typedef __attribute__((ext_vector_type(16))) __bf16        v16bf;
typedef __attribute__((ext_vector_type(8)))  float         v8f;
typedef __attribute__((ext_vector_type(8)))  unsigned short us8;
typedef __attribute__((ext_vector_type(4)))  float         f32x4;
typedef __attribute__((ext_vector_type(4)))  int           i32x4;

union ABf16 { v16bf bf; us8 h[2]; };

static __device__ __forceinline__ v8f vzero8() {
  v8f z = {0.f, 0.f, 0.f, 0.f, 0.f, 0.f, 0.f, 0.f};
  return z;
}

// float -> bf16 via native convert (RNE), bit pattern as ushort
static __device__ __forceinline__ unsigned short f2bf(float f) {
  union { __bf16 b; unsigned short u; } cv;
  cv.b = (__bf16)f;
  return cv.u;
}

// convert 8 consecutive f32 -> 8 bf16
static __device__ __forceinline__ us8 cvt8(const float* p) {
  f32x4 x0 = *(const f32x4*)(p);
  f32x4 x1 = *(const f32x4*)(p + 4);
  us8 o;
  o[0] = f2bf(x0[0]); o[1] = f2bf(x0[1]); o[2] = f2bf(x0[2]); o[3] = f2bf(x0[3]);
  o[4] = f2bf(x1[0]); o[5] = f2bf(x1[1]); o[6] = f2bf(x1[2]); o[7] = f2bf(x1[3]);
  return o;
}

static __device__ __forceinline__ v8f wmma_bf16(const ABf16& a, const ABf16& b, v8f c) {
  return __builtin_amdgcn_wmma_f32_16x16x32_bf16(false, a.bf, false, b.bf,
                                                 (short)0, c, false, false);
}

// =====================================================================
// K0: LDS-tiled transpose + convert weight to bf16  (WT[n][k] = bf16(W[k][n]))
// 32x32 tile per block, 256 threads (4 elements each); both sides coalesced.
// =====================================================================
__global__ __launch_bounds__(256) void mha_wt_kernel(const float* __restrict__ W,
                                                     unsigned short* __restrict__ WT) {
  __shared__ float tile[32][33];
  const int tx = threadIdx.x & 31;       // column within tile
  const int ty = threadIdx.x >> 5;       // 8 row-groups of 4
  const int bx = blockIdx.x & 15;        // DMODEL/32 = 16 tile cols (n)
  const int by = blockIdx.x >> 4;        // 16 tile rows (k)

#pragma unroll
  for (int i = 0; i < 4; ++i) {
    int k = by * 32 + ty * 4 + i;
    tile[ty * 4 + i][tx] = W[(size_t)k * DMODEL + bx * 32 + tx];
  }
  __syncthreads();
#pragma unroll
  for (int i = 0; i < 4; ++i) {
    int n = bx * 32 + ty * 4 + i;
    WT[(size_t)n * DMODEL + by * 32 + tx] = f2bf(tile[tx][ty * 4 + i]);
  }
}

// =====================================================================
// K1: projection GEMM  Y = X @ W + b.
// One wave computes a 16x64 strip (4 col tiles).  Double-buffered K loop:
// next A/B tiles are in flight while current WMMAs execute.
//   vmode 0 : write head-split   dst[B,H,S,DEPTH]   (Q, K)
//   vmode 1 : write transposed   dst[B,H,DEPTH,S]   (V)
// =====================================================================
__global__ __launch_bounds__(256) void mha_qkv_kernel(
    const float* __restrict__ X, const unsigned short* __restrict__ WT,
    const float* __restrict__ bias, unsigned short* __restrict__ dst, int vmode) {
  const int wv   = threadIdx.x >> 5;
  const int lane = threadIdx.x & 31;
  const int half = lane >> 4;
  const int ln   = lane & 15;

  const int g     = blockIdx.x * 8 + wv;     // global strip id
  const int colt  = g & 7;                   // DMODEL/64 strips
  const int rowt  = g >> 3;                  // BS/16 row tiles
  const int rbase = rowt * 16;
  const int cbase = colt * 64;

  const float* xrow = X + (size_t)(rbase + ln) * DMODEL;
  const unsigned short* wbase = WT + (size_t)(cbase + ln) * DMODEL;

  v8f acc[4] = {vzero8(), vzero8(), vzero8(), vzero8()};

  const int NK = DMODEL / 32;
  ABf16 a_cur, b_cur[4];
  a_cur.h[0] = cvt8(xrow + half * 16);
  a_cur.h[1] = cvt8(xrow + half * 16 + 8);
#pragma unroll
  for (int j = 0; j < 4; ++j) {
    const us8* wp = (const us8*)(wbase + (size_t)j * 16 * DMODEL + half * 16);
    b_cur[j].h[0] = wp[0];
    b_cur[j].h[1] = wp[1];
  }

#pragma unroll 2
  for (int t = 0; t < NK - 1; ++t) {
    const int kn = (t + 1) * 32;
    ABf16 a_nxt, b_nxt[4];
    a_nxt.h[0] = cvt8(xrow + kn + half * 16);
    a_nxt.h[1] = cvt8(xrow + kn + half * 16 + 8);
#pragma unroll
    for (int j = 0; j < 4; ++j) {
      const us8* wp = (const us8*)(wbase + (size_t)j * 16 * DMODEL + kn + half * 16);
      b_nxt[j].h[0] = wp[0];
      b_nxt[j].h[1] = wp[1];
    }
#pragma unroll
    for (int j = 0; j < 4; ++j) acc[j] = wmma_bf16(a_cur, b_cur[j], acc[j]);
    a_cur = a_nxt;
#pragma unroll
    for (int j = 0; j < 4; ++j) b_cur[j] = b_nxt[j];
  }
#pragma unroll
  for (int j = 0; j < 4; ++j) acc[j] = wmma_bf16(a_cur, b_cur[j], acc[j]);

  const int b  = rbase / SEQ;                // rows of a tile share b (SEQ%16==0)
  const int s0 = rbase & (SEQ - 1);
  const int h  = cbase >> 6;                 // 64-wide strip => single head

#pragma unroll
  for (int j = 0; j < 4; ++j) {
    const int   dd   = j * 16 + ln;          // depth index within head
    const float bval = bias[cbase + dd];
    if (vmode == 0) {
      size_t base = (((size_t)(b * NHEAD + h) * SEQ) + s0) * DEPTH + dd;
#pragma unroll
      for (int r = 0; r < 8; ++r) {
        int m = half * 8 + r;
        dst[base + (size_t)m * DEPTH] = f2bf(acc[j][r] + bval);
      }
    } else {
      size_t base = (((size_t)(b * NHEAD + h) * DEPTH) + dd) * SEQ + s0 + half * 8;
      us8 o;
#pragma unroll
      for (int r = 0; r < 8; ++r) o[r] = f2bf(acc[j][r] + bval);
      *(us8*)(dst + base) = o;
    }
  }
}

// =====================================================================
// K2: attention. One workgroup (8 waves) per (b, h, 16-row q block).
// =====================================================================
#define SCS 516   // f32 scores LDS row stride (x4B multiple of 16)
#define ATS 544   // bf16 attn  LDS row stride (x2B multiple of 16)

__global__ __launch_bounds__(256) void mha_attn_kernel(
    const unsigned short* __restrict__ Qh,   // [B,H,S,DEPTH] bf16
    const unsigned short* __restrict__ Kh,   // [B,H,S,DEPTH] bf16
    const unsigned short* __restrict__ VhT,  // [B,H,DEPTH,S] bf16
    const int* __restrict__ mask, const float* __restrict__ adjoin,
    const float* __restrict__ dist,
    float* __restrict__ attn_out,            // [B,H,S,S] f32
    unsigned short* __restrict__ ctx) {      // [BS, DMODEL] bf16
  __shared__ __align__(16) float          sc[16 * SCS];
  __shared__ __align__(16) unsigned short at[16 * ATS];
  __shared__ float redA[256];
  __shared__ float redB[256];
  __shared__ float outred[4 * 16 * 16];

  const int tid  = threadIdx.x;
  const int wv   = tid >> 5;
  const int lane = tid & 31;
  const int half = lane >> 4;
  const int ln   = lane & 15;

  const int bh    = blockIdx.x >> 5;         // b*H + h
  const int qt    = blockIdx.x & 31;
  const int b     = bh >> 3;                 // / NHEAD
  const int h     = bh & 7;
  const int qbase = qt * 16;

  // V operand rows this block will need in phase C; prefetch them now
  // (emits global_prefetch_b8) so they arrive under phases A/B.
  const int dt = wv & 3;     // depth tile 0..3 (16 cols each)
  const int kh = wv >> 2;    // K half: 0 -> [0,256), 1 -> [256,512)
  const unsigned short* vrow = VhT + ((size_t)bh * DEPTH + dt * 16 + ln) * SEQ;
  if (half == 0) {
#pragma unroll
    for (int c = 0; c < 4; ++c)
      __builtin_prefetch(vrow + kh * 256 + c * 64, 0, 3);
  }

  // -------- phase A: raw scores Q K^T -> LDS.  All A/B tiles preloaded, then
  // the 8-WMMA chain runs against in-flight loads.
  const unsigned short* qrow = Qh + ((size_t)bh * SEQ + qbase + ln) * DEPTH;
  v8f acc[4] = {vzero8(), vzero8(), vzero8(), vzero8()};

  ABf16 a[2], bm[2][4];
#pragma unroll
  for (int kk2 = 0; kk2 < 2; ++kk2) {
    const int kk = kk2 * 32;
    a[kk2].h[0] = *(const us8*)(qrow + kk + half * 16);
    a[kk2].h[1] = *(const us8*)(qrow + kk + half * 16 + 8);
#pragma unroll
    for (int j = 0; j < 4; ++j) {
      int cb = (wv * 4 + j) * 16;
      const unsigned short* krow = Kh + ((size_t)bh * SEQ + cb + ln) * DEPTH;
      const us8* kp = (const us8*)(krow + kk + half * 16);
      bm[kk2][j].h[0] = kp[0];
      bm[kk2][j].h[1] = kp[1];
    }
  }
#pragma unroll
  for (int kk2 = 0; kk2 < 2; ++kk2)
#pragma unroll
    for (int j = 0; j < 4; ++j)
      acc[j] = wmma_bf16(a[kk2], bm[kk2][j], acc[j]);

#pragma unroll
  for (int j = 0; j < 4; ++j) {
    int key = (wv * 4 + j) * 16 + ln;
#pragma unroll
    for (int r = 0; r < 8; ++r) {
      int m = half * 8 + r;
      sc[m * SCS + key] = acc[j][r];
    }
  }
  __syncthreads();

  // -------- phase B: fused elementwise (coalesced vector loads) + softmax ----
  {
    const float RESC_NUM = 3.7182818f;       // 1 + e
    int r   = tid >> 4;                      // row 0..15
    int seg = tid & 15;                      // 16 segments of 32 columns
    float* row = sc + r * SCS + seg * 32;
    size_t eb = ((size_t)b * SEQ + qbase + r) * SEQ + seg * 32;

    float mx = -1.0e30f;
#pragma unroll
    for (int c = 0; c < 32; c += 4) {
      f32x4 s  = *(f32x4*)(row + c);
      f32x4 wd = __builtin_nontemporal_load((const f32x4*)(dist + eb + c));
      i32x4 mk = __builtin_nontemporal_load((const i32x4*)(mask + eb + c));
      f32x4 aj = __builtin_nontemporal_load((const f32x4*)(adjoin + eb + c));
#pragma unroll
      for (int i = 0; i < 4; ++i) {
        float t = fmaxf(s[i], 0.0f);
        t = t * (RESC_NUM / (1.0f + expf(1.0f - wd[i]))) * 0.125f;
        t += (float)mk[i] * -1.0e9f;
        t += aj[i];
        s[i] = t;
        mx = fmaxf(mx, t);
      }
      *(f32x4*)(row + c) = s;
    }
    redA[tid] = mx;
    __syncthreads();
    float rowmax = -1.0e30f;
#pragma unroll
    for (int i = 0; i < 16; ++i) rowmax = fmaxf(rowmax, redA[(r << 4) + i]);

    float lsum = 0.0f;
#pragma unroll 4
    for (int c = 0; c < 32; ++c) {
      float e = expf(row[c] - rowmax);
      row[c] = e;
      lsum += e;
    }
    redB[tid] = lsum;
    __syncthreads();
    float rsum = 0.0f;
#pragma unroll
    for (int i = 0; i < 16; ++i) rsum += redB[(r << 4) + i];
    float inv = 1.0f / rsum;

    float* gout = attn_out + ((size_t)bh * SEQ + qbase + r) * SEQ + seg * 32;
    unsigned short* arow = at + r * ATS + seg * 32;
#pragma unroll
    for (int c = 0; c < 32; c += 4) {
      f32x4 p;
      p[0] = row[c + 0] * inv;
      p[1] = row[c + 1] * inv;
      p[2] = row[c + 2] * inv;
      p[3] = row[c + 3] * inv;
      __builtin_nontemporal_store(p, (f32x4*)(gout + c));  // 268MB stream > L2
      arow[c + 0] = f2bf(p[0]);
      arow[c + 1] = f2bf(p[1]);
      arow[c + 2] = f2bf(p[2]);
      arow[c + 3] = f2bf(p[3]);
    }
  }
  __syncthreads();

  // -------- phase C: ctx = attn @ V (K split across wave pairs), V double-buffered
  const int kk0 = kh * 256;
  v8f oacc = vzero8();

  ABf16 v_cur;
  {
    const us8* vp = (const us8*)(vrow + kk0 + half * 16);
    v_cur.h[0] = vp[0];
    v_cur.h[1] = vp[1];
  }
#pragma unroll
  for (int t = 0; t < 7; ++t) {
    const int kk = kk0 + t * 32;
    ABf16 v_nxt;
    {
      const us8* vp = (const us8*)(vrow + kk + 32 + half * 16);
      v_nxt.h[0] = vp[0];
      v_nxt.h[1] = vp[1];
    }
    ABf16 aa;
    aa.h[0] = *(const us8*)(at + ln * ATS + kk + half * 16);
    aa.h[1] = *(const us8*)(at + ln * ATS + kk + half * 16 + 8);
    oacc = wmma_bf16(aa, v_cur, oacc);
    v_cur = v_nxt;
  }
  {
    const int kk = kk0 + 7 * 32;
    ABf16 aa;
    aa.h[0] = *(const us8*)(at + ln * ATS + kk + half * 16);
    aa.h[1] = *(const us8*)(at + ln * ATS + kk + half * 16 + 8);
    oacc = wmma_bf16(aa, v_cur, oacc);
  }

  if (kh == 1) {
#pragma unroll
    for (int r = 0; r < 8; ++r)
      outred[(dt * 16 + half * 8 + r) * 16 + ln] = oacc[r];
  }
  __syncthreads();
  if (kh == 0) {
    int    col  = h * DEPTH + dt * 16 + ln;
    size_t base = ((size_t)b * SEQ + qbase) * DMODEL + col;
#pragma unroll
    for (int r = 0; r < 8; ++r) {
      int m = half * 8 + r;
      float v = oacc[r] + outred[(dt * 16 + m) * 16 + ln];
      ctx[base + (size_t)m * DMODEL] = f2bf(v);
    }
  }
}

// =====================================================================
// K3: output projection  out = ctx(bf16) @ Wo + bo  -> f32 [BS, DMODEL]
// 16x64 strip per wave, double-buffered K loop.
// =====================================================================
__global__ __launch_bounds__(256) void mha_outproj_kernel(
    const unsigned short* __restrict__ ctx, const unsigned short* __restrict__ WoT,
    const float* __restrict__ bo, float* __restrict__ out) {
  const int wv   = threadIdx.x >> 5;
  const int lane = threadIdx.x & 31;
  const int half = lane >> 4;
  const int ln   = lane & 15;

  const int g     = blockIdx.x * 8 + wv;
  const int colt  = g & 7;
  const int rowt  = g >> 3;
  const int rbase = rowt * 16;
  const int cbase = colt * 64;

  const unsigned short* arow  = ctx + (size_t)(rbase + ln) * DMODEL;
  const unsigned short* wbase = WoT + (size_t)(cbase + ln) * DMODEL;

  v8f acc[4] = {vzero8(), vzero8(), vzero8(), vzero8()};

  const int NK = DMODEL / 32;
  ABf16 a_cur, b_cur[4];
  a_cur.h[0] = *(const us8*)(arow + half * 16);
  a_cur.h[1] = *(const us8*)(arow + half * 16 + 8);
#pragma unroll
  for (int j = 0; j < 4; ++j) {
    const us8* wp = (const us8*)(wbase + (size_t)j * 16 * DMODEL + half * 16);
    b_cur[j].h[0] = wp[0];
    b_cur[j].h[1] = wp[1];
  }

#pragma unroll 2
  for (int t = 0; t < NK - 1; ++t) {
    const int kn = (t + 1) * 32;
    ABf16 a_nxt, b_nxt[4];
    a_nxt.h[0] = *(const us8*)(arow + kn + half * 16);
    a_nxt.h[1] = *(const us8*)(arow + kn + half * 16 + 8);
#pragma unroll
    for (int j = 0; j < 4; ++j) {
      const us8* wp = (const us8*)(wbase + (size_t)j * 16 * DMODEL + kn + half * 16);
      b_nxt[j].h[0] = wp[0];
      b_nxt[j].h[1] = wp[1];
    }
#pragma unroll
    for (int j = 0; j < 4; ++j) acc[j] = wmma_bf16(a_cur, b_cur[j], acc[j]);
    a_cur = a_nxt;
#pragma unroll
    for (int j = 0; j < 4; ++j) b_cur[j] = b_nxt[j];
  }
#pragma unroll
  for (int j = 0; j < 4; ++j) acc[j] = wmma_bf16(a_cur, b_cur[j], acc[j]);

#pragma unroll
  for (int j = 0; j < 4; ++j) {
    const float bval = bo[cbase + j * 16 + ln];
    size_t base = (size_t)rbase * DMODEL + cbase + j * 16 + ln;
#pragma unroll
    for (int r = 0; r < 8; ++r) {
      int m = half * 8 + r;
      out[base + (size_t)m * DMODEL] = acc[j][r] + bval;
    }
  }
}

// =====================================================================
// launch
// =====================================================================
extern "C" void kernel_launch(void* const* d_in, const int* in_sizes, int n_in,
                              void* d_out, int out_size, void* d_ws, size_t ws_size,
                              hipStream_t stream) {
  const float* q      = (const float*)d_in[0];
  const float* k      = (const float*)d_in[1];
  const float* v      = (const float*)d_in[2];
  const int*   mask   = (const int*)d_in[3];
  const float* adjoin = (const float*)d_in[4];
  const float* dist   = (const float*)d_in[5];
  const float* Wq     = (const float*)d_in[6];
  const float* bq     = (const float*)d_in[7];
  const float* Wk     = (const float*)d_in[8];
  const float* bk     = (const float*)d_in[9];
  const float* Wv     = (const float*)d_in[10];
  const float* bv     = (const float*)d_in[11];
  const float* Wo     = (const float*)d_in[12];
  const float* bo     = (const float*)d_in[13];

  float* out  = (float*)d_out;                       // [BS, D]
  float* attn = out + (size_t)BS * DMODEL;           // [B,H,S,S]

  // workspace layout (bf16 = ushort): 4 x W^T (0.5MB each) + Qh/Kh/VhT/ctx (16MB each)
  unsigned short* WqT = (unsigned short*)d_ws;
  unsigned short* WkT = WqT + (size_t)DMODEL * DMODEL;
  unsigned short* WvT = WkT + (size_t)DMODEL * DMODEL;
  unsigned short* WoT = WvT + (size_t)DMODEL * DMODEL;
  unsigned short* Qh  = WoT + (size_t)DMODEL * DMODEL;
  unsigned short* Kh  = Qh + (size_t)BS * DMODEL;
  unsigned short* VhT = Kh + (size_t)BS * DMODEL;
  unsigned short* ctx = VhT + (size_t)BS * DMODEL;

  dim3 blk(256);

  const int wtBlocks = (DMODEL / 32) * (DMODEL / 32);  // 256
  mha_wt_kernel<<<wtBlocks, blk, 0, stream>>>(Wq, WqT);
  mha_wt_kernel<<<wtBlocks, blk, 0, stream>>>(Wk, WkT);
  mha_wt_kernel<<<wtBlocks, blk, 0, stream>>>(Wv, WvT);
  mha_wt_kernel<<<wtBlocks, blk, 0, stream>>>(Wo, WoT);

  const int gemmBlocks = (BS / 16) * (DMODEL / 64) / 8;  // 1024
  mha_qkv_kernel<<<gemmBlocks, blk, 0, stream>>>(q, WqT, bq, Qh, 0);
  mha_qkv_kernel<<<gemmBlocks, blk, 0, stream>>>(k, WkT, bk, Kh, 0);
  mha_qkv_kernel<<<gemmBlocks, blk, 0, stream>>>(v, WvT, bv, VhT, 1);

  const int attnBlocks = BATCH * NHEAD * (SEQ / 16);     // 8192
  mha_attn_kernel<<<attnBlocks, blk, 0, stream>>>(Qh, Kh, VhT, mask, adjoin, dist,
                                                  attn, ctx);

  mha_outproj_kernel<<<gemmBlocks, blk, 0, stream>>>(ctx, WoT, bo, out);
}
